// MultiHeadSelfAttention_49125835931998
// MI455X (gfx1250) — matrix operand order repaired and checked by reading, not compiled, not run on hardware
//
#include <hip/hip_runtime.h>
#include <hip/hip_bf16.h>

typedef __attribute__((ext_vector_type(16))) __bf16 bf16x16;
typedef __attribute__((ext_vector_type(8)))  __bf16 bf16x8;
typedef __attribute__((ext_vector_type(8)))  float  f32x8;
typedef __attribute__((ext_vector_type(4)))  unsigned int u32x4;
typedef __attribute__((ext_vector_type(8)))  int i32x8;
typedef __attribute__((ext_vector_type(4)))  int i32x4;

typedef unsigned short u16;
typedef unsigned int   u32;

#define S_LEN   2048
#define B_SZ    2
#define D_DIM   1024
#define H_NUM   16
#define HD      64
#define N_ROWS  (B_SZ * S_LEN)   // 4096

#if defined(__HIP_DEVICE_COMPILE__) && \
    __has_builtin(__builtin_amdgcn_tensor_load_to_lds) && \
    __has_builtin(__builtin_amdgcn_s_wait_tensorcnt)
#define HAVE_TDM 1
#if __has_include(<hip/amd_detail/amd_gfx1250_TDM.h>)
#define TDM_ARITY6 1
#endif
#endif

__device__ __forceinline__ u16 f32_to_bf16(float f) {
  u32 u = __builtin_bit_cast(u32, f);
  u = (u + 0x7FFFu + ((u >> 16) & 1u)) >> 16;   // round-to-nearest-even
  return (u16)u;
}

__device__ __forceinline__ f32x8 wmma_bf16(bf16x16 a, bf16x16 b, f32x8 c) {
  return __builtin_amdgcn_wmma_f32_16x16x32_bf16(false, a, false, b, (short)0, c,
                                                 false, false);
}

__device__ __forceinline__ bf16x16 cat8(bf16x8 lo, bf16x8 hi) {
  return __builtin_shufflevector(lo, hi, 0, 1, 2, 3, 4, 5, 6, 7,
                                         8, 9, 10, 11, 12, 13, 14, 15);
}

// A-fragment (16x32 bf16): lane holds row (lane&15); K = half*16+0..15
// (contiguous order, matching the B fragments).  rowptr = &src[row*ld + k0].
__device__ __forceinline__ bf16x16 load_a32(const u16* rowptr, int half) {
  bf16x8 lo = *(const bf16x8*)(rowptr + half * 16);
  bf16x8 hi = *(const bf16x8*)(rowptr + half * 16 + 8);
  return cat8(lo, hi);
}

// ---------------------------------------------------------------- converts
__global__ __launch_bounds__(256) void cvt_bf16_kernel(
    const float* __restrict__ src, u16* __restrict__ dst, int n) {
  int i = blockIdx.x * 256 + threadIdx.x;
  if (i < n) dst[i] = f32_to_bf16(src[i]);
}

// wt[c][r] = bf16(w[r][c])  (1024x1024)
__global__ __launch_bounds__(256) void cvt_transpose_kernel(
    const float* __restrict__ w, u16* __restrict__ wt) {
  int i = blockIdx.x * 256 + threadIdx.x;
  int r = i >> 10, c = i & 1023;
  wt[(size_t)c * D_DIM + r] = f32_to_bf16(w[i]);
}

// ---------------------------------------------------------------- GEMM
// Block = 8 waves: one 128(M) x 64(N) tile of  Y = A[4096x1024] * W + bias.
// Each wave owns a 16x64 sub-tile.  Per 64-wide K-step the shared 64(N)x64(K)
// bf16 weight tile (8 KB) is staged in LDS, double buffered, via the Tensor
// Data Mover (TENSORcnt); all 16 B-fragment ds loads are issued before the
// 8-WMMA burst so the matrix pipe runs back-to-back.  A-fragments are
// software-pipelined from global one K-step ahead.
// mode 0: write bf16 head-split [B,H,S,Hd]; mode 1: bf16 [B,H,Hd,S];
// mode 2: fp32 row-major [4096,1024].
__global__ __launch_bounds__(256) void gemm_bf16_kernel(
    const u16* __restrict__ A, const u16* __restrict__ Bt,
    const float* __restrict__ bias, u16* __restrict__ outH,
    float* __restrict__ outF, int mode) {
  __shared__ __align__(64) u16 bstage[2][64 * 64];   // 2 x 8 KB

  const int tid = threadIdx.x;
  const int lane = tid & 31;
  const int wid = tid >> 5;
  const int mg = blockIdx.x >> 4;          // 0..31  (M groups of 128)
  const int nt = blockIdx.x & 15;          // 0..15  (N tiles of 64)
  const int m0 = (mg * 8 + wid) * 16;
  const int n0 = nt * 64;
  const int ln = lane & 15, half = lane >> 4;

  const u16* arow = A + (size_t)(m0 + ln) * D_DIM;

#if defined(HAVE_TDM)
  // One TDM 2-D descriptor: tile_dim0=64 (K, contiguous), tile_dim1=64 (N rows),
  // tensor row stride 1024 elements, data_size=2B.  ISA 8.3/8.4 bit packing.
  auto tdm_issue = [&](int buf, int k) {
    unsigned long long ga =
        (unsigned long long)(const void*)(Bt + (size_t)n0 * D_DIM + k);
    unsigned lds = (unsigned)(uintptr_t)(const void*)&bstage[buf][0];
    u32x4 g0 = { 1u,                                   // count=1, user mode
                 lds,                                  // lds_addr (bytes)
                 (u32)(ga & 0xFFFFFFFFu),
                 (u32)((ga >> 32) & 0x01FFFFFFu) | (2u << 30) };  // type=2
    const u32 td0 = D_DIM, td1 = D_DIM, tl0 = 64, tl1 = 64, st0 = D_DIM;
    i32x8 g1 = { (int)(1u << 16),                      // data_size=1 (2 bytes)
                 (int)((td0 & 0xFFFFu) << 16),         // tensor_dim0[15:0]
                 (int)((td0 >> 16) | ((td1 & 0xFFFFu) << 16)),
                 (int)((td1 >> 16) | (tl0 << 16)),     // tile_dim0
                 (int)(tl1),                           // tile_dim1, tile_dim2=0
                 (int)(st0),                           // tensor_dim0_stride lo32
                 0, 0 };
    i32x4 z4 = {0, 0, 0, 0};
#if defined(TDM_ARITY6)
    i32x8 z8 = {0, 0, 0, 0, 0, 0, 0, 0};
    __builtin_amdgcn_tensor_load_to_lds(g0, g1, z4, z4, z8, 0);
#else
    __builtin_amdgcn_tensor_load_to_lds(g0, g1, z4, z4, 0);
#endif
  };
#else
  // Fallback: cooperative staging, 256 threads x two b128 each.
  auto coop_stage = [&](int buf, int k) {
    int r = tid >> 2;              // 0..63 N row
    int c = (tid & 3) * 16;        // 0,16,32,48 within K
    *(bf16x8*)&bstage[buf][r * 64 + c] =
        *(const bf16x8*)(Bt + (size_t)(n0 + r) * D_DIM + k + c);
    *(bf16x8*)&bstage[buf][r * 64 + c + 8] =
        *(const bf16x8*)(Bt + (size_t)(n0 + r) * D_DIM + k + c + 8);
  };
#endif

  f32x8 acc[4] = {{}, {}, {}, {}};
  // One K-step: a0 covers k..k+31, a1 covers k+32..k+63.  Load all 8 B
  // fragments first, then fire 8 WMMAs back-to-back.
  auto compute = [&](int buf, bf16x16 a0, bf16x16 a1) {
    const u16* bb = &bstage[buf][0];
    bf16x16 bfr[8];
#pragma unroll
    for (int j = 0; j < 4; ++j) {
      const u16* col = bb + (j * 16 + ln) * 64 + half * 16;
      bfr[j]     = *(const bf16x16*)(col);
      bfr[4 + j] = *(const bf16x16*)(col + 32);
    }
#pragma unroll
    for (int j = 0; j < 4; ++j) acc[j] = wmma_bf16(a0, bfr[j], acc[j]);
#pragma unroll
    for (int j = 0; j < 4; ++j) acc[j] = wmma_bf16(a1, bfr[4 + j], acc[j]);
  };

  // prologue: stage K-step 0, preload first A fragment pair
#if defined(HAVE_TDM)
  if (wid == 0) tdm_issue(0, 0);
#else
  coop_stage(0, 0);
#endif
  bf16x16 a0_cur = load_a32(arow, half);
  bf16x16 a1_cur = load_a32(arow + 32, half);

  for (int i = 0; i < 15; ++i) {            // 16 K-steps of 64
#if defined(HAVE_TDM)
    if (wid == 0) {
      tdm_issue((i + 1) & 1, (i + 1) * 64);
      __builtin_amdgcn_s_wait_tensorcnt(1);   // oldest (buf i&1) complete
    }
#else
    coop_stage((i + 1) & 1, (i + 1) * 64);
#endif
    __syncthreads();
    bf16x16 a0_nxt = load_a32(arow + (i + 1) * 64, half);       // pipeline A
    bf16x16 a1_nxt = load_a32(arow + (i + 1) * 64 + 32, half);
    compute(i & 1, a0_cur, a1_cur);
    a0_cur = a0_nxt;
    a1_cur = a1_nxt;
    __syncthreads();            // all waves done with buf i&1 before reuse
  }
#if defined(HAVE_TDM)
  if (wid == 0) __builtin_amdgcn_s_wait_tensorcnt(0);
#endif
  __syncthreads();
  compute(1, a0_cur, a1_cur);   // K-step 15 lives in buffer 1

  // epilogue: bias + scatter per mode
#pragma unroll
  for (int j = 0; j < 4; ++j) {
    const int col = n0 + j * 16 + ln;
    const float bv = bias[col];
#pragma unroll
    for (int r = 0; r < 8; ++r) {
      float v = acc[j][r] + bv;
      int gm = m0 + r + half * 8;            // C layout: M = r + 8*half
      if (mode == 2) {
        outF[(size_t)gm * D_DIM + col] = v;
      } else {
        int b = gm >> 11, s = gm & 2047;
        int h = col >> 6, d = col & 63;
        size_t idx = (mode == 0)
            ? ((((size_t)b * H_NUM + h) * S_LEN + s) * HD + d)
            : ((((size_t)b * H_NUM + h) * HD + d) * S_LEN + s);
        outH[idx] = f32_to_bf16(v);
      }
    }
  }
}

// ---------------------------------------------------------------- flash attn
// One wave: 16 query rows of one (b,h), online softmax over all 2048 keys.
// K fragments are preloaded one block ahead; V fragments are issued before the
// softmax VALU section so global latency is hidden behind it.
__global__ __launch_bounds__(256) void flash_attn_kernel(
    const u16* __restrict__ Q, const u16* __restrict__ Km,
    const u16* __restrict__ Vt, u16* __restrict__ O) {
  __shared__ __align__(32) u16 pbuf[8][16 * 32];  // per-wave P staging
  const int lane = threadIdx.x & 31;
  const int wid = threadIdx.x >> 5;
  const int gw = blockIdx.x * 8 + wid;            // 0..4095
  const int bh = gw >> 7;                          // 0..31  (B*H)
  const int q0 = (gw & 127) * 16;
  const int ln = lane & 15, half = lane >> 4;
  const float kscale = 0.18033688011112042f;       // log2(e) / sqrt(64)

  const u16* qrow = Q + ((size_t)bh * S_LEN + q0 + ln) * HD;
  bf16x16 qf0 = load_a32(qrow + 0, half);
  bf16x16 qf1 = load_a32(qrow + 32, half);

  const u16* kbase = Km + (size_t)bh * S_LEN * HD + (size_t)ln * HD + half * 16;
  const u16* vbase = Vt + (size_t)bh * HD * S_LEN + (size_t)ln * S_LEN + half * 16;
  u16* pb = pbuf[wid];

  f32x8 o0 = {}, o1 = {}, o2 = {}, o3 = {};
  float mrun[8], lrun[8];
#pragma unroll
  for (int r = 0; r < 8; ++r) { mrun[r] = -3.0e38f; lrun[r] = 0.0f; }

  // preload K fragments for the first key block
  bf16x16 k00 = *(const bf16x16*)(kbase + 0);
  bf16x16 k01 = *(const bf16x16*)(kbase + 32);
  bf16x16 k10 = *(const bf16x16*)(kbase + 16 * HD);
  bf16x16 k11 = *(const bf16x16*)(kbase + 16 * HD + 32);

  for (int kb = 0; kb < S_LEN; kb += 32) {
    f32x8 s0 = {}, s1 = {};
    s0 = wmma_bf16(qf0, k00, s0);
    s0 = wmma_bf16(qf1, k01, s0);
    s1 = wmma_bf16(qf0, k10, s1);
    s1 = wmma_bf16(qf1, k11, s1);

    // issue V fragment loads + next K fragment loads early (hidden by softmax)
    const u16* v0 = vbase + kb;
    bf16x16 vf0 = *(const bf16x16*)(v0);
    bf16x16 vf1 = *(const bf16x16*)(v0 + 16 * S_LEN);
    bf16x16 vf2 = *(const bf16x16*)(v0 + 32 * S_LEN);
    bf16x16 vf3 = *(const bf16x16*)(v0 + 48 * S_LEN);
    if (kb + 32 < S_LEN) {
      const u16* kn = kbase + (size_t)(kb + 32) * HD;
      __builtin_prefetch(kn + 32 * HD, 0, 1);
      k00 = *(const bf16x16*)(kn + 0);
      k01 = *(const bf16x16*)(kn + 32);
      k10 = *(const bf16x16*)(kn + 16 * HD);
      k11 = *(const bf16x16*)(kn + 16 * HD + 32);
    }

#pragma unroll
    for (int r = 0; r < 8; ++r) { s0[r] *= kscale; s1[r] *= kscale; }

    // row max across 32 keys (elementwise, then across 16 lanes of each half)
    float bm[8];
#pragma unroll
    for (int r = 0; r < 8; ++r) bm[r] = fmaxf(s0[r], s1[r]);
#pragma unroll
    for (int msk = 1; msk <= 8; msk <<= 1)
#pragma unroll
      for (int r = 0; r < 8; ++r)
        bm[r] = fmaxf(bm[r], __shfl_xor(bm[r], msk, 32));

    float alpha[8];
#pragma unroll
    for (int r = 0; r < 8; ++r) {
      float mn = fmaxf(mrun[r], bm[r]);
      alpha[r] = exp2f(mrun[r] - mn);
      mrun[r] = mn;
    }

    f32x8 p0 = {}, p1 = {};
#pragma unroll
    for (int r = 0; r < 8; ++r) {
      p0[r] = exp2f(s0[r] - mrun[r]);
      p1[r] = exp2f(s1[r] - mrun[r]);
    }

    float bs[8];
#pragma unroll
    for (int r = 0; r < 8; ++r) bs[r] = p0[r] + p1[r];
#pragma unroll
    for (int msk = 1; msk <= 8; msk <<= 1)
#pragma unroll
      for (int r = 0; r < 8; ++r) bs[r] += __shfl_xor(bs[r], msk, 32);

#pragma unroll
    for (int r = 0; r < 8; ++r) {
      lrun[r] = lrun[r] * alpha[r] + bs[r];
      o0[r] *= alpha[r];
      o1[r] *= alpha[r];
      o2[r] *= alpha[r];
      o3[r] *= alpha[r];
    }

    // P (C layout) -> LDS row-major 16x32, reload as A-fragment.
    // DS ops are in-order per wave (DScnt); no barrier needed within a wave.
#pragma unroll
    for (int r = 0; r < 8; ++r) {
      pb[(r + 8 * half) * 32 + ln]      = f32_to_bf16(p0[r]);
      pb[(r + 8 * half) * 32 + 16 + ln] = f32_to_bf16(p1[r]);
    }
    asm volatile("" ::: "memory");  // keep IR from reordering LDS ld past st
    bf16x16 pf = load_a32(pb + ln * 32, half);

    o0 = wmma_bf16(pf, vf0, o0);
    o1 = wmma_bf16(pf, vf1, o1);
    o2 = wmma_bf16(pf, vf2, o2);
    o3 = wmma_bf16(pf, vf3, o3);
  }

  // normalize rows and write bf16 attention output [B*S, 1024]
  const int b = bh >> 4, h = bh & 15;
#pragma unroll
  for (int r = 0; r < 8; ++r) {
    float inv = 1.0f / lrun[r];
    int grow = b * S_LEN + q0 + r + 8 * half;
    u16* orow = O + (size_t)grow * D_DIM + h * HD;
    orow[0 * 16 + ln] = f32_to_bf16(o0[r] * inv);
    orow[1 * 16 + ln] = f32_to_bf16(o1[r] * inv);
    orow[2 * 16 + ln] = f32_to_bf16(o2[r] * inv);
    orow[3 * 16 + ln] = f32_to_bf16(o3[r] * inv);
  }
}

// ---------------------------------------------------------------- launcher
extern "C" void kernel_launch(void* const* d_in, const int* in_sizes, int n_in,
                              void* d_out, int out_size, void* d_ws,
                              size_t ws_size, hipStream_t stream) {
  const float* x  = (const float*)d_in[0];
  const float* wq = (const float*)d_in[1];
  const float* bq = (const float*)d_in[2];
  const float* wk = (const float*)d_in[3];
  const float* bk = (const float*)d_in[4];
  const float* wv = (const float*)d_in[5];
  const float* bv = (const float*)d_in[6];
  const float* wo = (const float*)d_in[7];
  const float* bo = (const float*)d_in[8];
  float* out = (float*)d_out;

  const size_t ND = (size_t)N_ROWS * D_DIM;   // 4M elems
  const size_t DD = (size_t)D_DIM * D_DIM;    // 1M elems
  u16* xb  = (u16*)d_ws;
  u16* wqT = xb + ND;
  u16* wkT = wqT + DD;
  u16* wvT = wkT + DD;
  u16* woT = wvT + DD;
  u16* Qb  = woT + DD;   // [B,H,S,Hd]
  u16* Kb  = Qb + ND;    // [B,H,S,Hd]
  u16* Vtb = Kb + ND;    // [B,H,Hd,S]
  u16* Ob  = Vtb + ND;   // [B*S, D]
  // total: 24M u16 = 48 MiB of workspace

  cvt_bf16_kernel<<<(int)(ND / 256), 256, 0, stream>>>(x, xb, (int)ND);
  cvt_transpose_kernel<<<(int)(DD / 256), 256, 0, stream>>>(wq, wqT);
  cvt_transpose_kernel<<<(int)(DD / 256), 256, 0, stream>>>(wk, wkT);
  cvt_transpose_kernel<<<(int)(DD / 256), 256, 0, stream>>>(wv, wvT);
  cvt_transpose_kernel<<<(int)(DD / 256), 256, 0, stream>>>(wo, woT);

  gemm_bf16_kernel<<<512, 256, 0, stream>>>(xb, wqT, bq, Qb, nullptr, 0);
  gemm_bf16_kernel<<<512, 256, 0, stream>>>(xb, wkT, bk, Kb, nullptr, 0);
  gemm_bf16_kernel<<<512, 256, 0, stream>>>(xb, wvT, bv, Vtb, nullptr, 1);

  flash_attn_kernel<<<512, 256, 0, stream>>>(Qb, Kb, Vtb, Ob);

  gemm_bf16_kernel<<<512, 256, 0, stream>>>(Ob, woT, bo, nullptr, out, 2);
}
